// Head_67654324847418
// MI455X (gfx1250) — matrix operand order repaired and checked by reading, not compiled, not run on hardware
//
#include <hip/hip_runtime.h>
#include <hip/hip_bf16.h>

typedef __attribute__((ext_vector_type(16))) _Float16 v16h;
typedef __attribute__((ext_vector_type(4)))  _Float16 v4h;
typedef __attribute__((ext_vector_type(2)))  _Float16 v2h;
typedef __attribute__((ext_vector_type(8)))  float    v8f;

#define B_  16
#define T_  2048
#define C_  200
#define H_  64
#define KT  32            // key tile
#define WAVES 8
#define QW  16            // query rows per wave
#define QB  (WAVES*QW)    // 128 query rows per block
#define NEG_BIG (-60000.0f)   // f16-representable "-inf" for masked keys
#define LOG2E 1.4426950408889634f

// Column permutation converting a row-major 32-wide h-block into the packed
// per-lane element order of the WMMA 16-bit A operand (ISA 7.12.2).
__device__ __forceinline__ int apermute(int h) {
    int o = (h >> 3) & 3;
    return h + ((o == 1) ? 8 : (o == 2) ? -8 : 0);
}

__device__ __forceinline__ v2h shfl_xor_v2h(v2h v, int s) {
    union { v2h h; int i; } u;
    u.h = v;
    u.i = __shfl_xor(u.i, s, 32);
    return u.h;
}

// ---------------- projection: q,k,v = x @ W{q,k,v} ------------------------------
// q: f16, row-major [B,T,H], scale (1/sqrt(H))*log2(e) folded in so the
//    attention kernel can use the raw hardware exp2 (softmax in base 2).
// k: f16, row-major [B,T,H].
// v: f16, TRANSPOSED [B,H,T] so the P@V B-operand is contiguous along keys.
__global__ __launch_bounds__(256) void head_proj_kernel(
    const float* __restrict__ x,
    const float* __restrict__ Wk, const float* __restrict__ Wq, const float* __restrict__ Wv,
    _Float16* __restrict__ qg, _Float16* __restrict__ kg, _Float16* __restrict__ vgT)
{
    __shared__ _Float16 sv[4][H_];
    const int tid = threadIdx.x;
    const int h   = tid & (H_ - 1);
    const int rl  = tid >> 6;
    const long row0 = (long)blockIdx.x * 4;
    const long row  = row0 + rl;                 // row in [0, B*T)
    const float* xr = x + row * C_;
    float aq = 0.f, ak = 0.f, av = 0.f;
#pragma unroll 8
    for (int c = 0; c < C_; ++c) {
        float xv = xr[c];
        ak = fmaf(xv, Wk[c * H_ + h], ak);
        aq = fmaf(xv, Wq[c * H_ + h], aq);
        av = fmaf(xv, Wv[c * H_ + h], av);
    }
    qg[(size_t)row * H_ + h] = (_Float16)(aq * (0.125f * LOG2E));
    kg[(size_t)row * H_ + h]           = (_Float16)ak;
    sv[rl][h] = (_Float16)av;
    __syncthreads();
    if (tid < H_) {                              // transpose 4x64 -> 64x4, 8B stores
        int b  = (int)(row0 / T_);
        int t0 = (int)(row0 % T_);
        v4h w = { sv[0][tid], sv[1][tid], sv[2][tid], sv[3][tid] };
        *(v4h*)(vgT + ((size_t)b * H_ + tid) * T_ + t0) = w;
    }
}

// ---------------- flash attention: fp32 softmax/accum, f16 WMMA ----------------
__global__ __launch_bounds__(256) void head_attn_kernel(
    const _Float16* __restrict__ qg, const _Float16* __restrict__ kg,
    const _Float16* __restrict__ vgT, const int* __restrict__ maskp,
    float* __restrict__ out)
{
    __shared__ _Float16 Pst[WAVES][QW * KT];    // per-wave P stage, packed A order

    const int tid  = threadIdx.x;
    const int wave = tid >> 5;
    const int lane = tid & 31;
    const int m    = lane & 15;
    const int hi   = lane >> 4;
    const int b     = blockIdx.x / (T_ / QB);
    const int qbase = (blockIdx.x % (T_ / QB)) * QB;
    const int qrow0 = qbase + wave * QW;

    // A operand direct from global: row m, packed positions hi*16..hi*16+15
    const _Float16* qrow = qg + ((size_t)b * T_ + qrow0 + m) * H_ + hi * 16;
    const v16h a0 = *(const v16h*)(qrow);        // h-block [0,32)
    const v16h a1 = *(const v16h*)(qrow + 32);   // h-block [32,64)

    v16h ones;                                   // B operand of all-ones: row sums via WMMA
#pragma unroll
    for (int e = 0; e < 16; ++e) ones[e] = (_Float16)1.0f;

    v8f o0 = {}, o1 = {}, o2 = {}, o3 = {};      // O accumulator, 16x64 fp32
    float mrow[8], lrow[8];
#pragma unroll
    for (int r = 0; r < 8; ++r) { mrow[r] = -3.0e38f; lrow[r] = 0.f; }

    const int* mb = maskp + b * T_;
    const _Float16* kbase = kg  + (size_t)b * T_ * H_;
    const _Float16* vbase = vgT + (size_t)b * H_ * T_;
    // P store positions in contiguous key order (col0 = m, col1 = 16+m)
    const int pos0 = m;
    const int pos1 = m + 16;
    _Float16* Pw = &Pst[wave][0];

    for (int kb = 0; kb < T_; kb += KT) {
        // ---- S = Q @ K^T : B operands contiguous 32B per lane from global ----
        const _Float16* kr0 = kbase + (size_t)(kb + m) * H_ + hi * 16;
        const _Float16* kr1 = kbase + (size_t)(kb + 16 + m) * H_ + hi * 16;
        if (kb + KT < T_) {                      // hint next key tile toward cache
            __builtin_prefetch(kbase + (size_t)(kb + KT + m) * H_, 0, 3);
            __builtin_prefetch(vbase + (size_t)m * T_ + kb + KT, 0, 3);
        }
        v8f s0 = {}, s1 = {};
        s0 = __builtin_amdgcn_wmma_f32_16x16x32_f16(false, a0, false, *(const v16h*)(kr0),      (short)0, s0, false, false);
        s0 = __builtin_amdgcn_wmma_f32_16x16x32_f16(false, a1, false, *(const v16h*)(kr0 + 32), (short)0, s0, false, false);
        s1 = __builtin_amdgcn_wmma_f32_16x16x32_f16(false, a0, false, *(const v16h*)(kr1),      (short)0, s1, false, false);
        s1 = __builtin_amdgcn_wmma_f32_16x16x32_f16(false, a1, false, *(const v16h*)(kr1 + 32), (short)0, s1, false, false);

        // ---- column padding mask as additive bias (2 cndmask instead of 16) ----
        const float bias0 = mb[kb + m]      ? 0.0f : NEG_BIG;
        const float bias1 = mb[kb + 16 + m] ? 0.0f : NEG_BIG;
        float v0[8], v1[8];
#pragma unroll
        for (int r = 0; r < 8; ++r) { v0[r] = s0[r] + bias0; v1[r] = s1[r] + bias1; }

        // ---- row max across 16 lanes: packed-f16 butterfly (stabilizer only) ----
        v2h pk[4];
#pragma unroll
        for (int j = 0; j < 4; ++j) {
            v2h t;
            t[0] = (_Float16)fmaxf(v0[2*j],     v1[2*j]);
            t[1] = (_Float16)fmaxf(v0[2*j + 1], v1[2*j + 1]);
            pk[j] = t;
        }
#pragma unroll
        for (int s = 1; s < 16; s <<= 1)
#pragma unroll
            for (int j = 0; j < 4; ++j)
                pk[j] = __builtin_elementwise_max(pk[j], shfl_xor_v2h(pk[j], s));

        // ---- online update: raw v_exp_f32 (base-2 domain, no libm fixup) ----
#pragma unroll
        for (int r = 0; r < 8; ++r) {
            float mx    = (float)pk[r >> 1][r & 1];
            float mnew  = fmaxf(mrow[r], mx);
            float alpha = __builtin_amdgcn_exp2f(mrow[r] - mnew);
            float p0 = __builtin_amdgcn_exp2f(v0[r] - mnew);
            float p1 = __builtin_amdgcn_exp2f(v1[r] - mnew);
            mrow[r] = mnew;
            lrow[r] *= alpha;                     // sum added below via WMMA
            o0[r] *= alpha; o1[r] *= alpha; o2[r] *= alpha; o3[r] *= alpha;
            int prow = hi * 8 + r;
            Pw[prow * KT + pos0] = (_Float16)p0;
            Pw[prow * KT + pos1] = (_Float16)p1;
        }

        // ---- O += P @ V ; row sums of P via one extra WMMA against ones ----
        const v16h ap = *(const v16h*)(&Pw[m * KT + hi * 16]);
        v8f zc = {};
        v8f lsum = __builtin_amdgcn_wmma_f32_16x16x32_f16(false, ap, false, ones, (short)0, zc, false, false);
        const _Float16* vr = vbase + (size_t)m * T_ + kb + hi * 16;
        o0 = __builtin_amdgcn_wmma_f32_16x16x32_f16(false, ap, false, *(const v16h*)(vr),           (short)0, o0, false, false);
        o1 = __builtin_amdgcn_wmma_f32_16x16x32_f16(false, ap, false, *(const v16h*)(vr + 16 * T_), (short)0, o1, false, false);
        o2 = __builtin_amdgcn_wmma_f32_16x16x32_f16(false, ap, false, *(const v16h*)(vr + 32 * T_), (short)0, o2, false, false);
        o3 = __builtin_amdgcn_wmma_f32_16x16x32_f16(false, ap, false, *(const v16h*)(vr + 48 * T_), (short)0, o3, false, false);
#pragma unroll
        for (int r = 0; r < 8; ++r) lrow[r] += lsum[r];
    }

    // ---- normalize, apply query-row mask, store fp32 ----
#pragma unroll
    for (int r = 0; r < 8; ++r) {
        int qrow_i = qrow0 + hi * 8 + r;
        float f = mb[qrow_i] ? (1.0f / lrow[r]) : 0.0f;
        size_t base = ((size_t)b * T_ + qrow_i) * H_ + m;
        out[base]      = o0[r] * f;
        out[base + 16] = o1[r] * f;
        out[base + 32] = o2[r] * f;
        out[base + 48] = o3[r] * f;
    }
}

extern "C" void kernel_launch(void* const* d_in, const int* in_sizes, int n_in,
                              void* d_out, int out_size, void* d_ws, size_t ws_size,
                              hipStream_t stream) {
    (void)in_sizes; (void)n_in; (void)out_size; (void)ws_size;
    const float* x    = (const float*)d_in[0];
    const int*   mask = (const int*)d_in[1];
    const float* Wk   = (const float*)d_in[2];
    const float* Wq   = (const float*)d_in[3];
    const float* Wv   = (const float*)d_in[4];
    float* out = (float*)d_out;

    const size_t N = (size_t)B_ * T_ * H_;
    _Float16* qg  = (_Float16*)d_ws;
    _Float16* kg  = qg + N;
    _Float16* vgT = kg + N;

    head_proj_kernel<<<(B_ * T_) / 4, 256, 0, stream>>>(x, Wk, Wq, Wv, qg, kg, vgT);
    head_attn_kernel<<<B_ * (T_ / QB), 256, 0, stream>>>(qg, kg, vgT, mask, out);
}
